// Head_68212670595336
// MI455X (gfx1250) — compile-verified, matrix operand order checked by reading, not run on hardware
//
#include <hip/hip_runtime.h>

// Problem constants (match reference)
#define BB 4
#define TT 4096
#define CCDIM 64
#define HSD 64
#define SCALE_QK 0.125f                   // C^-0.5 = 64^-0.5
#define ALIBI_SLOPE 0.70710678118654752f  // (2^8)^(-1/16)

typedef __attribute__((ext_vector_type(16))) __bf16 v16bf;
typedef __attribute__((ext_vector_type(8)))  float  v8f;

__device__ __forceinline__ v8f wmma_bf16(v16bf a, v16bf b, v8f c) {
  // D(f32 16x16) = A(bf16 16x32) * B(bf16 32x16) + C
  return __builtin_amdgcn_wmma_f32_16x16x32_bf16(
      /*neg_a=*/false, a, /*neg_b=*/false, b,
      /*c_mod=*/(short)0, c, /*reuse_a=*/false, /*reuse_b=*/false);
}

// K-index packing for 16-bit A fragments (ISA 7.12.2), assumed symmetric for B.
__device__ __forceinline__ int kpack(int j, int h) {
  return ((j >> 2) << 4) + (h << 3) + ((j & 3) << 1);
}

// Async global->LDS copy, 16B per lane (ASYNCcnt-tracked; ISA 10. / §15.18.3).
// LDS byte offset = low 32 bits of the flat address (LDS aperture truncation).
__device__ __forceinline__ void async_copy_b128(void* lds_dst, const void* src) {
  const unsigned lds_off = (unsigned)(uintptr_t)lds_dst;
  asm volatile("global_load_async_to_lds_b128 %0, %1, off"
               :: "v"(lds_off), "v"(src)
               : "memory");
}
__device__ __forceinline__ void wait_async0() {
  asm volatile("s_wait_asynccnt 0x0" ::: "memory");
}

// ---------------------------------------------------------------------------
// Projection: q/k/v = x @ W{q,k,v}, written as bf16 [B*T, 64] row-major.
// W matrices staged in LDS transposed (bf16) so every B fragment is two
// contiguous 16B ds loads instead of 16 strided global loads.
// ---------------------------------------------------------------------------
__global__ void __launch_bounds__(128)
proj_qkv(const float* __restrict__ x,
         const float* __restrict__ Wq, const float* __restrict__ Wk,
         const float* __restrict__ Wv,
         __bf16* __restrict__ qb, __bf16* __restrict__ kb,
         __bf16* __restrict__ vb) {
  __shared__ __align__(16) __bf16 wt[3][64 * 64];  // [w][col*64 + k] = W[k][col]

  const int tid  = threadIdx.x;
  const int wid  = tid >> 5;
  const int lane = tid & 31;
  const int lq   = lane & 15;
  const int h    = (lane >> 4) & 1;
  const int rowbase = blockIdx.x * 64 + wid * 16;   // row in [0, B*T)

  const float* Ws[3] = {Wq, Wk, Wv};
  __bf16*      Os[3] = {qb, kb, vb};

  // Stage all three W transposed into LDS (one coalesced pass per block).
  {
    const int r  = tid >> 1;          // input row k
    const int c0 = (tid & 1) * 32;    // output-col base
#pragma unroll
    for (int w = 0; w < 3; ++w) {
      const float* src = Ws[w] + (size_t)r * HSD + c0;
#pragma unroll
      for (int i = 0; i < 32; ++i)
        wt[w][(c0 + i) * 64 + r] = (__bf16)src[i];
    }
  }

  // A fragments from x rows (convert f32 -> bf16); row = rowbase + lq
  v16bf ax[2];
  const float* xrow = x + (size_t)(rowbase + lq) * CCDIM;
#pragma unroll
  for (int c = 0; c < 2; ++c) {
#pragma unroll
    for (int j = 0; j < 8; ++j) {
      const int k = c * 32 + kpack(j, h);
      ax[c][2 * j]     = (__bf16)xrow[k];
      ax[c][2 * j + 1] = (__bf16)xrow[k + 1];
    }
  }

  __syncthreads();

#pragma unroll
  for (int w = 0; w < 3; ++w) {
    __bf16* O = Os[w];
#pragma unroll
    for (int nt = 0; nt < 4; ++nt) {
      v8f acc = {};
#pragma unroll
      for (int c = 0; c < 2; ++c) {
        // B fragment from transposed-W LDS: pairs (k,k+1) are contiguous.
        const __bf16* wcol =
            &wt[w][(nt * 16 + lq) * 64 + c * 32 + (h << 3)];
        v16bf bw;
#pragma unroll
        for (int j = 0; j < 8; ++j) {
          const int k = ((j >> 2) << 4) + ((j & 3) << 1);  // 0..7, 16..23
          bw[2 * j]     = wcol[k];
          bw[2 * j + 1] = wcol[k + 1];
        }
        acc = wmma_bf16(ax[c], bw, acc);
      }
#pragma unroll
      for (int r = 0; r < 8; ++r) {
        const int row = rowbase + r + (h << 3);
        O[(size_t)row * HSD + nt * 16 + lq] = (__bf16)acc[r];
      }
    }
  }
}

// ---------------------------------------------------------------------------
// Flash attention (transposed): per wave, 16 queries; per block, 64 queries.
// Key loop in tiles of 32: S^T = K x Q^T ; online softmax ; O^T += V^T x P^T.
// K tile staged via async global->LDS DMA; V transposed through registers.
// ---------------------------------------------------------------------------
__global__ void __launch_bounds__(128)
attn_fwd(const __bf16* __restrict__ qbuf, const __bf16* __restrict__ kbuf,
         const __bf16* __restrict__ vbuf, float* __restrict__ out) {
  __shared__ __align__(16) __bf16 kts[32 * HSD];   // [key][hd]  row-major
  __shared__ __align__(16) __bf16 vts[HSD * 32];   // [hd][key]  transposed

  const int tid  = threadIdx.x;
  const int wid  = tid >> 5;
  const int lane = tid & 31;
  const int lq   = lane & 15;
  const int h    = (lane >> 4) & 1;
  const int b    = blockIdx.y;
  const int qblk = blockIdx.x * 64;
  const int q0   = qblk + wid * 16;                // wave's query-tile base
  const size_t bofs = (size_t)b * TT * HSD;

  // B fragments of Q^T (hd x 16 queries); per-lane reads one Q row (query=lq).
  v16bf bqf[2];
  {
    const __bf16* qrow = qbuf + bofs + (size_t)(q0 + lq) * HSD;
#pragma unroll
    for (int c = 0; c < 2; ++c)
#pragma unroll
      for (int j = 0; j < 8; ++j) {
        const int k = c * 32 + kpack(j, h);
        bqf[c][2 * j]     = qrow[k];
        bqf[c][2 * j + 1] = qrow[k + 1];
      }
  }

  v8f o[4] = {};                 // O^T accumulators: 4 hd-tiles x (16x16)
  float m = -1e30f, l = 0.0f;    // per-query softmax state (query = lane&15)

  const int kend = qblk + 63;    // causal bound for this block
  const int skey = tid & 31;
  const int shb  = (tid >> 5) << 4;   // 0,16,32,48

  for (int kb = 0; kb <= kend; kb += 32) {
    __syncthreads();             // previous-iteration LDS reads done
    {
      // K tile: async DMA straight into LDS (row-major), 2 x 16B per thread.
      const __bf16* ksrc = kbuf + bofs + (size_t)(kb + skey) * HSD + shb;
      async_copy_b128(&kts[skey * HSD + shb], ksrc);
      async_copy_b128(&kts[skey * HSD + shb + 8], ksrc + 8);
      // V tile: transpose through registers into LDS.
      const __bf16* vsrc = vbuf + bofs + (size_t)(kb + skey) * HSD + shb;
#pragma unroll
      for (int i = 0; i < 16; ++i) vts[(shb + i) * 32 + skey] = vsrc[i];
      // Prefetch next tile (global_prefetch_b8).
      if (kb + 32 <= kend) {
        __builtin_prefetch(kbuf + bofs + (size_t)(kb + 32 + skey) * HSD, 0, 0);
        __builtin_prefetch(vbuf + bofs + (size_t)(kb + 32 + skey) * HSD, 0, 0);
      }
      wait_async0();             // our async writes to LDS have landed
    }
    __syncthreads();

    if (kb <= q0 + 15) {         // wave-uniform causal skip
      // S^T tiles: (16 keys x 16 queries) x 2 key-subtiles
      v8f s[2] = {};
#pragma unroll
      for (int t = 0; t < 2; ++t) {
        const __bf16* krow = &kts[(t * 16 + lq) * HSD];
#pragma unroll
        for (int c = 0; c < 2; ++c) {
          v16bf ak;
#pragma unroll
          for (int j = 0; j < 8; ++j) {
            const int k = c * 32 + kpack(j, h);
            ak[2 * j]     = krow[k];
            ak[2 * j + 1] = krow[k + 1];
          }
          s[t] = wmma_bf16(ak, bqf[c], s[t]);
        }
      }

      // Scale + ALiBi + causal mask; per-lane keys: {8h..8h+7} U {16+8h..+7}
      const int qg = q0 + lq;
      float p[2][8];
      float mloc = -1e30f;
#pragma unroll
      for (int t = 0; t < 2; ++t)
#pragma unroll
        for (int r = 0; r < 8; ++r) {
          const int kg = kb + t * 16 + r + (h << 3);
          float v = s[t][r] * SCALE_QK + (float)(kg - qg) * ALIBI_SLOPE;
          v = (kg <= qg) ? v : -1e30f;
          p[t][r] = v;
          mloc = fmaxf(mloc, v);
        }
      mloc = fmaxf(mloc, __shfl_xor(mloc, 16));
      const float mnew  = fmaxf(m, mloc);
      const float alpha = __expf(m - mnew);
      float lloc = 0.0f;
#pragma unroll
      for (int t = 0; t < 2; ++t)
#pragma unroll
        for (int r = 0; r < 8; ++r) {
          const float e = __expf(p[t][r] - mnew);
          p[t][r] = e;
          lloc += e;
        }
      lloc += __shfl_xor(lloc, 16);
      l = l * alpha + lloc;
      m = mnew;
#pragma unroll
      for (int tm = 0; tm < 4; ++tm)
#pragma unroll
        for (int r = 0; r < 8; ++r) o[tm][r] *= alpha;

      // P^T B-fragment: pure in-register repack of S^T C-fragments
      v16bf bp;
#pragma unroll
      for (int j = 0; j < 8; ++j) {
        bp[2 * j]     = (__bf16)p[j >> 2][(j & 3) << 1];
        bp[2 * j + 1] = (__bf16)p[j >> 2][((j & 3) << 1) + 1];
      }

      // O^T += V^T x P^T (4 hd-tiles)
#pragma unroll
      for (int tm = 0; tm < 4; ++tm) {
        v16bf av;
        const __bf16* vrow = &vts[(tm * 16 + lq) * 32];
#pragma unroll
        for (int j = 0; j < 8; ++j) {
          const int k = kpack(j, h);
          av[2 * j]     = vrow[k];
          av[2 * j + 1] = vrow[k + 1];
        }
        o[tm] = wmma_bf16(av, bp, o[tm]);
      }
    }
  }

  // Normalize and write: O^T element (hd = tm*16 + r + 8h, query = lq)
  const float inv = 1.0f / l;
  float* orow = out + bofs + (size_t)(q0 + lq) * HSD;
#pragma unroll
  for (int tm = 0; tm < 4; ++tm)
#pragma unroll
    for (int r = 0; r < 8; ++r)
      orow[tm * 16 + r + (h << 3)] = o[tm][r] * inv;
}

// ---------------------------------------------------------------------------
extern "C" void kernel_launch(void* const* d_in, const int* in_sizes, int n_in,
                              void* d_out, int out_size, void* d_ws,
                              size_t ws_size, hipStream_t stream) {
  const float* x  = (const float*)d_in[0];
  const float* Wq = (const float*)d_in[1];
  const float* Wk = (const float*)d_in[2];
  const float* Wv = (const float*)d_in[3];
  float* out = (float*)d_out;

  // Workspace: q, k, v as bf16 [B*T, 64] each (6 MB total)
  __bf16* qb = (__bf16*)d_ws;
  __bf16* kb = qb + (size_t)BB * TT * HSD;
  __bf16* vb = kb + (size_t)BB * TT * HSD;

  proj_qkv<<<dim3((BB * TT) / 64), dim3(128), 0, stream>>>(x, Wq, Wk, Wv,
                                                           qb, kb, vb);
  attn_fwd<<<dim3(TT / 64, BB), dim3(128), 0, stream>>>(qb, kb, vb, out);
}
